// NormalNN_9345848836280
// MI455X (gfx1250) — compile-verified, matrix operand order checked by reading, not run on hardware
//
#include <hip/hip_runtime.h>

// ---------------- problem constants ----------------
#define NN   100000
#define FF   500
#define HH   64
#define EE   3200000
#define KK   10
#define CLSN 40
#define ALPHA_STRIDE (KK + 1)   // alpha is [HID, K+1] row-major

typedef __attribute__((ext_vector_type(16))) __bf16 v16bf;
typedef __attribute__((ext_vector_type(8)))  float  v8f;

#if defined(__HIP_DEVICE_COMPILE__) && defined(__gfx1250__)
#define USE_ASYNC_LDS 1
#endif

// ---- gfx1250 async global->LDS copy (per-lane b128), with host/fallback path
__device__ __forceinline__ void async_g2l_b128(float* lds_ptr, const float* gptr)
{
#ifdef USE_ASYNC_LDS
    // generic LDS address: low 32 bits are the LDS byte offset
    unsigned loff = (unsigned)(unsigned long long)lds_ptr;
    asm volatile("global_load_async_to_lds_b128 %0, %1, off"
                 :: "v"(loff), "v"(gptr) : "memory");
#else
    *(float4*)lds_ptr = *(const float4*)gptr;
#endif
}

__device__ __forceinline__ void async_wait_all()
{
#ifdef USE_ASYNC_LDS
    asm volatile("s_wait_asynccnt 0x0" ::: "memory");
#endif
}

// ---- one 16x16x32 bf16 WMMA step from staged f32 LDS tiles
__device__ __forceinline__ v8f wmma_step(const float (*Alds)[36], const float (*Blds)[68],
                                         int wave, int lane, v8f acc)
{
    const int hlf  = lane >> 4;
    const int nloc = lane & 15;
    v16bf a, b;
#pragma unroll
    for (int i = 0; i < 16; ++i) {
        const int k = (i < 8 ? i : i + 8) + 8 * hlf;   // ISA 16-bit A/B layout
        a[i] = (__bf16)Alds[nloc][k];
        b[i] = (__bf16)Blds[k][wave * 16 + nloc];
    }
    return __builtin_amdgcn_wmma_f32_16x16x32_bf16(false, a, false, b,
                                                   (short)0, acc, false, false);
}

// =====================================================================
// fc1: X = relu(features @ W1 + b1) + noise*1e-5     (WMMA bf16)
// block = 128 threads = 4 waves; block computes 16 rows x 64 cols;
// wave w owns the 16-column tile w. A/B tiles staged via async b128
// global->LDS copies for the 15 full K-steps; masked tail for K=480..499.
// =====================================================================
__global__ __launch_bounds__(128) void k_fc1(const float* __restrict__ A,
                                             const float* __restrict__ W1,
                                             const float* __restrict__ b1,
                                             const float* __restrict__ noise,
                                             float* __restrict__ X)
{
    __shared__ float Alds[16][36];   // +4 pad: 16B-aligned rows, spread banks
    __shared__ float Blds[32][68];

    const int tid  = threadIdx.x;
    const int wave = tid >> 5;
    const int lane = tid & 31;
    const int row0 = blockIdx.x * 16;

    v8f acc = {};

    // ---- 15 full K-steps: async b128 staging, no bounds checks ----
    const int ar = tid >> 3;           // A row 0..15
    const int aq = (tid & 7) * 4;      // A k-quad 0,4,..,28
    for (int kb = 0; kb < 480; kb += 32) {
        async_g2l_b128(&Alds[ar][aq], &A[(size_t)(row0 + ar) * FF + kb + aq]);
#pragma unroll
        for (int j = 0; j < 4; ++j) {
            const int lin = tid + j * 128;       // 0..511 b128 chunks
            const int kk  = lin >> 4;            // B row 0..31
            const int nq  = (lin & 15) * 4;      // col quad
            async_g2l_b128(&Blds[kk][nq], &W1[(size_t)(kb + kk) * HH + nq]);
        }
        async_wait_all();
        __syncthreads();
        acc = wmma_step(Alds, Blds, wave, lane, acc);
        __syncthreads();
    }

    // ---- tail step kb=480 (20 valid K), zero-padded scalar staging ----
    for (int idx = tid; idx < 16 * 32; idx += 128) {
        const int r = idx >> 5, k = idx & 31;
        Alds[r][k] = (480 + k < FF) ? A[(size_t)(row0 + r) * FF + 480 + k] : 0.0f;
    }
    for (int idx = tid; idx < 32 * 64; idx += 128) {
        const int k = idx >> 6, n = idx & 63;
        Blds[k][n] = (480 + k < FF) ? W1[(size_t)(480 + k) * HH + n] : 0.0f;
    }
    __syncthreads();
    acc = wmma_step(Alds, Blds, wave, lane, acc);

    // ---- epilogue: C/D layout -> VGPR r holds row r + 8*half, col = lane&15
    const int hlf  = lane >> 4;
    const int nloc = lane & 15;
    const int col  = wave * 16 + nloc;
    const float bv = b1[col];
#pragma unroll
    for (int r = 0; r < 8; ++r) {
        const int row = row0 + r + 8 * hlf;
        float v = acc[r] + bv;
        v = v > 0.0f ? v : 0.0f;
        v += noise[(size_t)row * HH + col] * 1e-5f;
        X[(size_t)row * HH + col] = v;
    }
}

// =====================================================================
// small helpers
// =====================================================================
__global__ void k_zero(float* __restrict__ p, long long n)
{
    long long t = (long long)blockIdx.x * blockDim.x + threadIdx.x;
    if (t < n) p[t] = 0.0f;
}

// column sum of squares (64 columns) -> atomic partials
__global__ __launch_bounds__(256) void k_colsq(const float* __restrict__ X,
                                               float* __restrict__ out)
{
    __shared__ float red[256];
    const int c  = threadIdx.x & 63;
    const int rg = threadIdx.x >> 6;   // 0..3
    float s = 0.0f;
    for (long long r = (long long)blockIdx.x * 4 + rg; r < NN;
         r += (long long)gridDim.x * 4) {
        const float v = X[r * HH + c];
        s += v * v;
    }
    red[threadIdx.x] = s;
    __syncthreads();
    if (rg == 0)
        atomicAdd(&out[c], red[c] + red[64 + c] + red[128 + c] + red[192 + c]);
}

// sumsq -> reciprocal of clamped column norm (in place)
__global__ void k_finalize_norm(float* __restrict__ s)
{
    const int c = threadIdx.x;
    if (c < HH) {
        const float nrm = sqrtf(s[c]);
        s[c] = 1.0f / fmaxf(nrm, 1e-8f);
    }
}

// h0 = X * rnorm ; rst = alpha[:,0]*h0 ; second_last = 0   (X in place)
__global__ void k_scale_init(float* __restrict__ X, float* __restrict__ rst,
                             float* __restrict__ second,
                             const float* __restrict__ alpha,
                             const float* __restrict__ rn)
{
    long long t = (long long)blockIdx.x * blockDim.x + threadIdx.x;
    if (t >= (long long)NN * HH) return;
    const int c = (int)(t & 63);
    const float v = X[t] * rn[c];
    X[t]      = v;
    rst[t]    = alpha[c * ALPHA_STRIDE + 0] * v;
    second[t] = 0.0f;
}

// =====================================================================
// SpMM scatter: h_new[dst] += w * last_h[src]   (edge-parallel, f32 atomics;
// last_h (25.6MB) and h_new live in the 192MB L2)
// =====================================================================
__global__ __launch_bounds__(256) void k_spmm(const int* __restrict__ ei,
                                              const float* __restrict__ w,
                                              const float* __restrict__ hl,
                                              float* __restrict__ hn)
{
    long long t = (long long)blockIdx.x * blockDim.x + threadIdx.x;
    if (t >= (long long)EE * 16) return;
    const int e  = (int)(t >> 4);
    const int c0 = (int)(t & 15) * 4;
    const int s  = ei[e];
    const int d  = ei[EE + e];
    const float ww = w[e];
    const float4 v = *(const float4*)(hl + (size_t)s * HH + c0);
    float* p = hn + (size_t)d * HH + c0;
    atomicAdd(p + 0, ww * v.x);
    atomicAdd(p + 1, ww * v.y);
    atomicAdd(p + 2, ww * v.z);
    atomicAdd(p + 3, ww * v.w);
}

// dot1[c] = sum h*last, dot2[c] = sum h*second
__global__ __launch_bounds__(256) void k_dots(const float* __restrict__ h,
                                              const float* __restrict__ last,
                                              const float* __restrict__ second,
                                              float* __restrict__ scal)
{
    __shared__ float r1[256];
    __shared__ float r2[256];
    const int c  = threadIdx.x & 63;
    const int rg = threadIdx.x >> 6;
    float s1 = 0.0f, s2 = 0.0f;
    for (long long r = (long long)blockIdx.x * 4 + rg; r < NN;
         r += (long long)gridDim.x * 4) {
        const long long idx = r * HH + c;
        const float hv = h[idx];
        s1 += hv * last[idx];
        s2 += hv * second[idx];
    }
    r1[threadIdx.x] = s1;
    r2[threadIdx.x] = s2;
    __syncthreads();
    if (rg == 0) {
        atomicAdd(&scal[c],      r1[c] + r1[64 + c] + r1[128 + c] + r1[192 + c]);
        atomicAdd(&scal[64 + c], r2[c] + r2[64 + c] + r2[128 + c] + r2[192 + c]);
    }
}

// h -= dot1*last + dot2*second (in place), accumulate new sumsq
__global__ __launch_bounds__(256) void k_orth(float* __restrict__ h,
                                              const float* __restrict__ last,
                                              const float* __restrict__ second,
                                              const float* __restrict__ scal,
                                              float* __restrict__ sumsq)
{
    __shared__ float red[256];
    const int c  = threadIdx.x & 63;
    const int rg = threadIdx.x >> 6;
    const float d1 = scal[c];
    const float d2 = scal[64 + c];
    float s = 0.0f;
    for (long long r = (long long)blockIdx.x * 4 + rg; r < NN;
         r += (long long)gridDim.x * 4) {
        const long long idx = r * HH + c;
        const float v = h[idx] - d1 * last[idx] - d2 * second[idx];
        h[idx] = v;
        s += v * v;
    }
    red[threadIdx.x] = s;
    __syncthreads();
    if (rg == 0)
        atomicAdd(&sumsq[c], red[c] + red[64 + c] + red[128 + c] + red[192 + c]);
}

// h *= rnorm ; rst += alpha[:,it]*h
__global__ void k_scale_acc(float* __restrict__ h, float* __restrict__ rst,
                            const float* __restrict__ alpha,
                            const float* __restrict__ rn, int it)
{
    long long t = (long long)blockIdx.x * blockDim.x + threadIdx.x;
    if (t >= (long long)NN * HH) return;
    const int c = (int)(t & 63);
    const float v = h[t] * rn[c];
    h[t] = v;
    rst[t] += alpha[c * ALPHA_STRIDE + it] * v;
}

// =====================================================================
// fc2: out = rst @ W2 + b2  (WMMA bf16; N=40 -> 3 column tiles, last masked)
// block = 96 threads = 3 waves, 16 rows per block
// =====================================================================
__global__ __launch_bounds__(96) void k_fc2(const float* __restrict__ rst,
                                            const float* __restrict__ W2,
                                            const float* __restrict__ b2,
                                            float* __restrict__ out)
{
    __shared__ float Rlds[16][65];
    const int tid  = threadIdx.x;
    const int wave = tid >> 5;
    const int lane = tid & 31;
    const int row0 = blockIdx.x * 16;
    const int hlf  = lane >> 4;
    const int nloc = lane & 15;
    const int col  = wave * 16 + nloc;

    for (int idx = tid; idx < 16 * 64; idx += 96) {
        const int r = idx >> 6, k = idx & 63;
        Rlds[r][k] = rst[(size_t)(row0 + r) * HH + k];
    }
    __syncthreads();

    v8f acc = {};
    for (int kb = 0; kb < 64; kb += 32) {
        v16bf a, b;
#pragma unroll
        for (int i = 0; i < 16; ++i) {
            const int k = (i < 8 ? i : i + 8) + 8 * hlf;
            a[i] = (__bf16)Rlds[nloc][kb + k];
            b[i] = (col < CLSN) ? (__bf16)W2[(size_t)(kb + k) * CLSN + col]
                                : (__bf16)0.0f;
        }
        acc = __builtin_amdgcn_wmma_f32_16x16x32_bf16(false, a, false, b,
                                                      (short)0, acc, false, false);
    }

    if (col < CLSN) {
        const float bv = b2[col];
#pragma unroll
        for (int r = 0; r < 8; ++r) {
            const int row = row0 + r + 8 * hlf;
            out[(size_t)row * CLSN + col] = acc[r] + bv;
        }
    }
}

// =====================================================================
// host launcher
// =====================================================================
extern "C" void kernel_launch(void* const* d_in, const int* in_sizes, int n_in,
                              void* d_out, int out_size, void* d_ws, size_t ws_size,
                              hipStream_t stream)
{
    const float* features = (const float*)d_in[0];
    const float* noise    = (const float*)d_in[1];
    const float* norm_A   = (const float*)d_in[2];
    const float* W1       = (const float*)d_in[3];
    const float* b1       = (const float*)d_in[4];
    const float* W2       = (const float*)d_in[5];
    const float* b2       = (const float*)d_in[6];
    const float* alpha    = (const float*)d_in[7];
    const int*   ei       = (const int*)d_in[8];
    float* out = (float*)d_out;

    const long long NH = (long long)NN * HH;   // 6.4M
    float* ws   = (float*)d_ws;
    float* bufA = ws;                 // last_h   (starts as X / h0)
    float* bufB = ws + NH;            // h_new
    float* bufC = ws + 2 * NH;        // second_last
    float* rst  = ws + 3 * NH;
    float* scal = ws + 4 * NH;        // [0:64) dot1, [64:128) dot2, [128:192) sumsq->1/norm

    const int elemBlocks = (int)((NH + 255) / 256);        // 25000
    const int spmmBlocks = (int)(((long long)EE * 16 + 255) / 256);

    // --- fc1 + column normalize ---
    k_fc1<<<NN / 16, 128, 0, stream>>>(features, W1, b1, noise, bufA);
    k_zero<<<1, 256, 0, stream>>>(scal, 192);
    k_colsq<<<512, 256, 0, stream>>>(bufA, scal + 128);
    k_finalize_norm<<<1, 64, 0, stream>>>(scal + 128);
    k_scale_init<<<elemBlocks, 256, 0, stream>>>(bufA, rst, bufC, alpha, scal + 128);

    // --- K rounds of normal-basis conv ---
    float* last   = bufA;
    float* second = bufC;
    float* hnew   = bufB;
    for (int it = 1; it <= KK; ++it) {
        k_zero<<<elemBlocks, 256, 0, stream>>>(hnew, NH);
        k_spmm<<<spmmBlocks, 256, 0, stream>>>(ei, norm_A, last, hnew);
        k_zero<<<1, 256, 0, stream>>>(scal, 192);
        k_dots<<<512, 256, 0, stream>>>(hnew, last, second, scal);
        k_orth<<<512, 256, 0, stream>>>(hnew, last, second, scal, scal + 128);
        k_finalize_norm<<<1, 64, 0, stream>>>(scal + 128);
        k_scale_acc<<<elemBlocks, 256, 0, stream>>>(hnew, rst, alpha, scal + 128, it);
        float* tmp = second;
        second = last;
        last   = hnew;
        hnew   = tmp;
    }

    // --- fc2 ---
    k_fc2<<<NN / 16, 96, 0, stream>>>(rst, W2, b2, out);
}
